// BNO_11227044511977
// MI455X (gfx1250) — compile-verified, matrix operand order checked by reading, not run on hardware
//
#include <hip/hip_runtime.h>

// ---------------------------------------------------------------------------
// Born-series FNO forward for gfx1250 (MI455X).
// All matmuls (dense layers AND the spectral conv, recast as truncated DFT
// matrix products) run through v_wmma_f32_16x16x32_f16 (f16 in, f32 accum).
// Dense per-pixel layers: b128 A streaming, B fragments moved to LDS with
// global_load_async_to_lds_b128 (ASYNCcnt), ds_load_b128 reads, and fully
// unrolled branch-free WMMA tiles (NT/KS are template constants).
// ---------------------------------------------------------------------------

typedef __attribute__((ext_vector_type(16))) _Float16 v16h;
typedef __attribute__((ext_vector_type(8)))  float    v8f;

#define NPIX 131072   // 2 * 256 * 256 padded pixels
#define NELT (NPIX*64)

__device__ __forceinline__ float gelu_f(float x){
  float x3 = x*x*x;
  float t = tanhf(0.7978845608028654f*(x + 0.044715f*x3));
  return 0.5f*x*(1.0f+t);
}

// K offset pattern within a 16-bit A/B fragment (per documented VGPR layout)
__device__ __forceinline__ int kpat(int v){ return (v<4) ? 2*v : 16 + 2*(v-4); }

// ---------------------------------------------------------------------------
// Specialized dense layer:  C[M,N] = act(A[M,K] x W[K,N] + bias), lda=K, ldc=N
// Template: NT = N/16 tiles, KS = K/32 steps  ->  branch-free unrolled WMMAs.
// 256 threads = 8 waves; block tile = 128 rows x N.
// ---------------------------------------------------------------------------
template<int NT, int KS>
__global__ __launch_bounds__(256)
void bno_dense_wmma(const float* __restrict__ A, const _Float16* __restrict__ Bt,
                    const float* __restrict__ bias, float* __restrict__ C,
                    int act)
{
  constexpr int N = NT*16;
  constexpr int K = KS*32;
  __shared__ _Float16 bsh[KS*NT*512];      // fragment-ordered B tiles
  const int tid  = threadIdx.x;
  const int wv   = tid >> 5;
  const int lane = tid & 31;
  const int half = lane >> 4;
  const int l15  = lane & 15;

  // ---- stage all B fragments into LDS via async-to-LDS (16B runs) ----
  // run layout: ridx = ((ks*NT + nt)*32 + ln)*2 + r ; each run = 8 f16 = 16B
  constexpr int totalRuns = KS*NT*64;      // multiple of 256 for all configs
  #pragma unroll
  for (int it = 0; it < totalRuns/256; ++it){
    int ridx = it*256 + tid;
    int r  = ridx & 1;
    int t  = ridx >> 1;
    int ln = t & 31;  t >>= 5;
    int nt = t % NT;
    int ks = t / NT;
    int n  = nt*16 + (ln & 15);
    int gk = ks*32 + (ln >> 4)*8 + r*16;
    const _Float16* gsrc = Bt + (size_t)n*K + gk;
    _Float16* ldst = bsh + ((size_t)(ks*NT + nt)*32 + ln)*16 + r*8;
    unsigned loff = (unsigned)(uintptr_t)ldst;          // LDS aperture: addr[31:0]
    unsigned long long ga = (unsigned long long)(uintptr_t)gsrc;
    asm volatile("global_load_async_to_lds_b128 %0, %1, off"
                 :: "v"(loff), "v"(ga) : "memory");
  }
  asm volatile("s_wait_asynccnt 0x0" ::: "memory");
  __syncthreads();

  const long row0 = (long)blockIdx.x*128 + wv*16;
  const float* ap = A + (row0 + l15)*(long)K + half*8;

  v8f zero = {0,0,0,0,0,0,0,0};
  v8f acc[NT];
  #pragma unroll
  for (int nt=0; nt<NT; ++nt) acc[nt] = zero;

  const v16h* bsv = (const v16h*)bsh;
  #pragma unroll
  for (int ks=0; ks<KS; ++ks){
    const float* aq = ap + ks*32;
    if (ks+1 < KS) __builtin_prefetch(aq + 32, 0, 0);   // global_prefetch_b8
    float4 f0 = *(const float4*)(aq);
    float4 f1 = *(const float4*)(aq + 4);
    float4 f2 = *(const float4*)(aq + 16);
    float4 f3 = *(const float4*)(aq + 20);
    v16h a;
    a[0]=(_Float16)f0.x;  a[1]=(_Float16)f0.y;  a[2]=(_Float16)f0.z;  a[3]=(_Float16)f0.w;
    a[4]=(_Float16)f1.x;  a[5]=(_Float16)f1.y;  a[6]=(_Float16)f1.z;  a[7]=(_Float16)f1.w;
    a[8]=(_Float16)f2.x;  a[9]=(_Float16)f2.y;  a[10]=(_Float16)f2.z; a[11]=(_Float16)f2.w;
    a[12]=(_Float16)f3.x; a[13]=(_Float16)f3.y; a[14]=(_Float16)f3.z; a[15]=(_Float16)f3.w;
    #pragma unroll
    for (int nt=0; nt<NT; ++nt){
      v16h b = bsv[(ks*NT + nt)*32 + lane];             // 2x ds_load_b128
      acc[nt] = __builtin_amdgcn_wmma_f32_16x16x32_f16(false, a, false, b, (short)0,
                                                       acc[nt], false, false);
    }
  }

  #pragma unroll
  for (int nt=0; nt<NT; ++nt){
    int col = nt*16 + l15;
    float bv = bias ? bias[col] : 0.0f;
    #pragma unroll
    for (int r=0; r<8; ++r){
      float v = acc[nt][r] + bv;
      if (act == 1) v = gelu_f(v);
      C[(row0 + half*8 + r)*(long)N + col] = v;
    }
  }
}

// ---------------------------------------------------------------------------
// Generic strided WMMA GEMM (spectral DFT stages, cropped proj_out):
// C[r,n] (+)= act( sgnA * sum_k A[r,k]*Bt[n,k] + bias[n] ),
// r = (r2*D1 + r1)*D0 + r0 with independent strides for A and C.
// ---------------------------------------------------------------------------
__global__ __launch_bounds__(32)
void bno_gemm_wmma(const float* __restrict__ A, const _Float16* __restrict__ Bt,
                   const float* __restrict__ bias, float* __restrict__ C,
                   int K, int D0, int D1,
                   long as0, long as1, long as2, long ask,
                   long cs0, long cs1, long cs2, long csn,
                   float sgnA, int accFlag, int act, int nMax)
{
  const int lane = threadIdx.x;
  const int half = lane >> 4;
  const int l15  = lane & 15;
  const int ncol = blockIdx.y*16 + l15;
  const _Float16* brow = Bt + (size_t)ncol * (size_t)K;

  int arow = blockIdx.x*16 + l15;
  int t0 = arow / D0, r0 = arow - t0*D0;
  int r2 = t0 / D1,  r1 = t0 - r2*D1;
  const long aBase = (long)r0*as0 + (long)r1*as1 + (long)r2*as2;

  long cAddr[8];
  v8f acc;
  #pragma unroll
  for (int r=0;r<8;++r){
    int row = blockIdx.x*16 + half*8 + r;
    int u0 = row / D0, q0 = row - u0*D0;
    int q2 = u0 / D1,  q1 = u0 - q2*D1;
    cAddr[r] = (long)q0*cs0 + (long)q1*cs1 + (long)q2*cs2 + (long)ncol*csn;
    acc[r] = (accFlag && ncol < nMax) ? C[cAddr[r]] : 0.0f;
  }

  const int kb0 = half*8;
  for (int k0=0; k0<K; k0+=32){
    v16h a, b;
    #pragma unroll
    for (int v=0; v<8; ++v){
      int kk = k0 + kb0 + kpat(v);
      a[2*v]   = (_Float16)(sgnA * A[aBase + (long)kk*ask]);
      a[2*v+1] = (_Float16)(sgnA * A[aBase + (long)(kk+1)*ask]);
      b[2*v]   = brow[kk];
      b[2*v+1] = brow[kk+1];
    }
    acc = __builtin_amdgcn_wmma_f32_16x16x32_f16(false, a, false, b, (short)0, acc,
                                                 false, false);
  }

  if (ncol < nMax){
    #pragma unroll
    for (int r=0;r<8;++r){
      float v = acc[r];
      if (bias) v += bias[ncol];
      if (act == 1) v = gelu_f(v);
      C[cAddr[r]] = v;
    }
  }
}

// ---------------------------------------------------------------------------
// Per-mode complex channel mix:  O[b,o] = sum_c X[b,c] * (kr + i ki)[c,o]
// X layout: (mode g, b, c) -> g*128 + b*64 + c.
// Weight layout: k[c][o][i][j] -> c*65536 + o*1024 + (i*32+j).
// Weights pre-scaled by 256 into f16 range; epilogue divides by 256.
// ---------------------------------------------------------------------------
__global__ __launch_bounds__(32)
void bno_modemul_wmma(const float* __restrict__ Xr, const float* __restrict__ Xi,
                      const float* __restrict__ Kr, const float* __restrict__ Ki,
                      float* __restrict__ Or, float* __restrict__ Oi, int gBase)
{
  const int lane = threadIdx.x;
  const int half = lane >> 4;
  const int l15  = lane & 15;
  const int z    = blockIdx.z;
  const int ncol = blockIdx.x*16 + l15;
  const long aOff = (long)(gBase + z) * 128;
  const float* Ar = Xr + aOff;
  const float* Ai = Xi + aOff;
  const float* Wr = Kr + z;
  const float* Wi = Ki + z;
  v8f cr = {0,0,0,0,0,0,0,0}, ci = {0,0,0,0,0,0,0,0};
  const int kb0  = half*8;
  const int arow = l15;
  for (int k0=0;k0<64;k0+=32){
    v16h ar, ai, am, br, bi;
    #pragma unroll
    for (int v=0;v<8;++v){
      int kk = k0 + kb0 + kpat(v);
      float r0v = (arow<2) ? Ar[arow*64 + kk]   : 0.f;
      float r1v = (arow<2) ? Ar[arow*64 + kk+1] : 0.f;
      float i0v = (arow<2) ? Ai[arow*64 + kk]   : 0.f;
      float i1v = (arow<2) ? Ai[arow*64 + kk+1] : 0.f;
      ar[2*v]=(_Float16)r0v;    ar[2*v+1]=(_Float16)r1v;
      ai[2*v]=(_Float16)i0v;    ai[2*v+1]=(_Float16)i1v;
      am[2*v]=(_Float16)(-i0v); am[2*v+1]=(_Float16)(-i1v);
      br[2*v]   = (_Float16)(Wr[(size_t)kk*65536     + (size_t)ncol*1024] * 256.f);
      br[2*v+1] = (_Float16)(Wr[(size_t)(kk+1)*65536 + (size_t)ncol*1024] * 256.f);
      bi[2*v]   = (_Float16)(Wi[(size_t)kk*65536     + (size_t)ncol*1024] * 256.f);
      bi[2*v+1] = (_Float16)(Wi[(size_t)(kk+1)*65536 + (size_t)ncol*1024] * 256.f);
    }
    cr = __builtin_amdgcn_wmma_f32_16x16x32_f16(false, ar, false, br, (short)0, cr, false, false);
    cr = __builtin_amdgcn_wmma_f32_16x16x32_f16(false, am, false, bi, (short)0, cr, false, false);
    ci = __builtin_amdgcn_wmma_f32_16x16x32_f16(false, ar, false, bi, (short)0, ci, false, false);
    ci = __builtin_amdgcn_wmma_f32_16x16x32_f16(false, ai, false, br, (short)0, ci, false, false);
  }
  #pragma unroll
  for (int r=0;r<8;++r){
    int row = half*8 + r;
    if (row < 2){
      Or[aOff + row*64 + ncol] = cr[r] * (1.0f/256.0f);
      Oi[aOff + row*64 + ncol] = ci[r] * (1.0f/256.0f);
    }
  }
}

// ---------------------------------------------------------------------------
// Truncated DFT matrices (f16, transposed [n][k]).  Forward matrices carry the
// 1/256 per-axis normalization; Gw carries the hermitian 2x weights.
// ---------------------------------------------------------------------------
__global__ void bno_init_dft(_Float16* fwr, _Float16* fwi,   // [j=32][w=256]
                             _Float16* fhr, _Float16* fhi,   // [m'=64][h=256]
                             _Float16* ghr, _Float16* ghi,   // [h=256][m'=64]
                             _Float16* gwr, _Float16* gwi)   // [w=256][j=32]
{
  int idx = blockIdx.x*blockDim.x + threadIdx.x;
  const float TWO_PI = 6.283185307179586f;
  if (idx < 32*256){
    int j = idx >> 8, w = idx & 255;
    float ang = TWO_PI * (float)((w*j) & 255) * (1.0f/256.0f);
    fwr[idx] = (_Float16)( cosf(ang) * (1.0f/256.0f));
    fwi[idx] = (_Float16)(-sinf(ang) * (1.0f/256.0f));
  }
  if (idx < 64*256){
    int mp = idx >> 8, h = idx & 255;
    int m = (mp < 32) ? mp : mp + 192;
    float ang = TWO_PI * (float)((h*m) & 255) * (1.0f/256.0f);
    fhr[idx] = (_Float16)( cosf(ang) * (1.0f/256.0f));
    fhi[idx] = (_Float16)(-sinf(ang) * (1.0f/256.0f));
  }
  if (idx < 256*64){
    int h = idx >> 6, mp = idx & 63;
    int m = (mp < 32) ? mp : mp + 192;
    float ang = TWO_PI * (float)((h*m) & 255) * (1.0f/256.0f);
    ghr[idx] = (_Float16)cosf(ang);
    ghi[idx] = (_Float16)sinf(ang);
  }
  if (idx < 256*32){
    int w = idx >> 5, j = idx & 31;
    float mult = (j==0) ? 1.0f : 2.0f;
    float ang = TWO_PI * (float)((w*j) & 255) * (1.0f/256.0f);
    gwr[idx] = (_Float16)(mult * cosf(ang));
    gwi[idx] = (_Float16)(mult * sinf(ang));
  }
}

// Convert f32 dense weight (cin x cout, [k][n]) -> f16 transposed [n][kPad], zero-padded.
__global__ void bno_conv_w(const float* __restrict__ w, _Float16* __restrict__ bt,
                           int cin, int cout, int kPad, int nPad)
{
  int idx = blockIdx.x*blockDim.x + threadIdx.x;
  if (idx >= kPad*nPad) return;
  int n = idx / kPad, k = idx - n*kPad;
  float v = (k < cin && n < cout) ? w[(size_t)k*cout + n] : 0.0f;
  bt[idx] = (_Float16)v;
}

// Build padded 5-channel (pad to 32) input features: [src, sos, pml, w/255, h/255, 0..]
__global__ void bno_pack_in(const float* __restrict__ sos, const float* __restrict__ pml,
                            const float* __restrict__ src, float* __restrict__ kin)
{
  int idx = blockIdx.x*blockDim.x + threadIdx.x;
  if (idx >= NPIX) return;
  int b = idx >> 16; int hw = idx & 65535; int h = hw >> 8; int w = hw & 255;
  bool in = (h < 224) && (w < 224);
  int si = (b*224 + h)*224 + w;
  float* o = kin + (size_t)idx*32;
  o[0] = in ? src[si] : 0.0f;
  o[1] = in ? sos[si] : 1.0f;
  o[2] = in ? pml[si] : 1.0f;
  o[3] = (float)w * (1.0f/255.0f);
  o[4] = (float)h * (1.0f/255.0f);
  #pragma unroll
  for (int c=5;c<32;++c) o[c] = 0.0f;
}

// s = L*u + Ms   (vectorized, in-place into s)
__global__ void bno_ew_s(const float4* __restrict__ L, const float4* __restrict__ u,
                         float4* __restrict__ s, int n4)
{
  int i = blockIdx.x*blockDim.x + threadIdx.x;
  if (i < n4){
    float4 a = L[i], b = u[i], c = s[i];
    c.x = a.x*b.x + c.x;  c.y = a.y*b.y + c.y;
    c.z = a.z*b.z + c.z;  c.w = a.w*b.w + c.w;
    s[i] = c;
  }
}

// born = Wu - G*(Hu - spec)   (vectorized)
__global__ void bno_ew_born(const float4* __restrict__ Wu, const float4* __restrict__ G,
                            const float4* __restrict__ Hu, const float4* __restrict__ spec,
                            float4* __restrict__ out, int n4)
{
  int i = blockIdx.x*blockDim.x + threadIdx.x;
  if (i < n4){
    float4 w = Wu[i], g = G[i], h = Hu[i], sp = spec[i], o;
    o.x = w.x - g.x*(h.x - sp.x);  o.y = w.y - g.y*(h.y - sp.y);
    o.z = w.z - g.z*(h.z - sp.z);  o.w = w.w - g.w*(h.w - sp.w);
    out[i] = o;
  }
}

// ---------------------------------------------------------------------------
// Host orchestration
// ---------------------------------------------------------------------------
extern "C" void kernel_launch(void* const* d_in, const int* in_sizes, int n_in,
                              void* d_out, int out_size, void* d_ws, size_t ws_size,
                              hipStream_t stream)
{
  (void)in_sizes; (void)n_in; (void)out_size; (void)ws_size;
  auto f = [&](int i)->const float* { return (const float*)d_in[i]; };

  const float* sos = f(0);
  const float* pml = f(1);
  const float* src = f(2);
  // proj_x: 3..6 (d1.w,d1.b,d2.w,d2.b); proj_k: 7..10
  // stages: base 11 + d*22: +0..3 L +4..7 G +8,9 u +10,11 Wu +12,13 Hu
  //   +14,15 Ms +16,17 out +18 k1r +19 k1i +20 k2r +21 k2i;  proj_out: 99..102

  char* wsb = (char*)d_ws; size_t off = 0;
  auto allocF = [&](size_t n)->float* {
    float* p = (float*)(wsb + off); off += n*sizeof(float); off = (off+255)&~(size_t)255; return p; };
  auto allocH = [&](size_t n)->_Float16* {
    _Float16* p = (_Float16*)(wsb + off); off += n*sizeof(_Float16); off = (off+255)&~(size_t)255; return p; };

  float* kin  = allocF((size_t)NPIX*32);
  float* kf   = allocF((size_t)NPIX*64);
  float* xb   = allocF((size_t)NPIX*64);
  float* t128 = allocF((size_t)NPIX*128);
  float* b1   = allocF((size_t)NPIX*64);
  float* b2   = allocF((size_t)NPIX*64);
  float* b3   = allocF((size_t)NPIX*64);
  float* b4   = allocF((size_t)NPIX*64);
  float* b5   = allocF((size_t)NPIX*64);
  float* X1r  = allocF(1048576); float* X1i = allocF(1048576);  // (b,h,c,j)
  float* X2r  = allocF(262144);  float* X2i = allocF(262144);   // (m',j,b,c)
  float* KKr  = allocF(262144);  float* KKi = allocF(262144);
  float* Y1r  = allocF(1048576); float* Y1i = allocF(1048576);  // (b,h,c,j)

  _Float16 *fwr = allocH(8192),  *fwi = allocH(8192);
  _Float16 *fhr = allocH(16384), *fhi = allocH(16384);
  _Float16 *ghr = allocH(16384), *ghi = allocH(16384);
  _Float16 *gwr = allocH(8192),  *gwi = allocH(8192);

  _Float16 *hPxd1 = allocH(32*128),  *hPxd2 = allocH(128*64);
  _Float16 *hPkd1 = allocH(32*128),  *hPkd2 = allocH(128*64);
  _Float16 *hLd1[4], *hLd2[4], *hGd1[4], *hGd2[4], *hU[4], *hWu[4], *hHu[4], *hMs[4], *hOut[4];
  for (int d=0; d<4; ++d){
    hLd1[d]=allocH(64*128); hLd2[d]=allocH(128*64);
    hGd1[d]=allocH(64*128); hGd2[d]=allocH(128*64);
    hU[d]=allocH(64*64); hWu[d]=allocH(64*64); hHu[d]=allocH(64*64);
    hMs[d]=allocH(64*64); hOut[d]=allocH(64*64);
  }
  _Float16 *hPod1 = allocH(64*128), *hPod2 = allocH(128*16);

  auto convW = [&](const float* s_, _Float16* d_, int cin, int cout, int kPad, int nPad){
    int tot = kPad*nPad;
    bno_conv_w<<<(tot+255)/256, 256, 0, stream>>>(s_, d_, cin, cout, kPad, nPad);
  };
  auto gemm = [&](const float* A, const _Float16* Bt, const float* bias, float* C,
                  int M, int N, int K, int D0, int D1,
                  long as0, long as1, long as2, long ask,
                  long cs0, long cs1, long cs2, long csn,
                  float sgnA, int acc, int act, int nMax){
    dim3 g(M/16, N/16, 1);
    bno_gemm_wmma<<<g, 32, 0, stream>>>(A, Bt, bias, C, K, D0, D1,
                                        as0,as1,as2,ask, cs0,cs1,cs2,csn,
                                        sgnA, acc, act, nMax);
  };
  // per-pixel dense layer (lda=K, ldc=N): static-shape dispatch
  auto dense = [&](const float* A, const _Float16* Bt, const float* bias,
                   float* C, int M, int N, int K, int act){
    dim3 g(M/128);
    if      (N==128 && K==32 ) bno_dense_wmma<8,1><<<g,256,0,stream>>>(A,Bt,bias,C,act);
    else if (N==64  && K==128) bno_dense_wmma<4,4><<<g,256,0,stream>>>(A,Bt,bias,C,act);
    else if (N==128 && K==64 ) bno_dense_wmma<8,2><<<g,256,0,stream>>>(A,Bt,bias,C,act);
    else if (N==64  && K==64 ) bno_dense_wmma<4,2><<<g,256,0,stream>>>(A,Bt,bias,C,act);
  };

  // --- per-call prep ---
  bno_init_dft<<<(16384+255)/256, 256, 0, stream>>>(fwr,fwi,fhr,fhi,ghr,ghi,gwr,gwi);
  convW(f(3),  hPxd1, 5,   128, 32, 128);
  convW(f(5),  hPxd2, 128, 64, 128, 64);
  convW(f(7),  hPkd1, 5,   128, 32, 128);
  convW(f(9),  hPkd2, 128, 64, 128, 64);
  for (int d=0; d<4; ++d){
    int s0 = 11 + d*22;
    convW(f(s0+0),  hLd1[d], 64,128, 64,128);
    convW(f(s0+2),  hLd2[d], 128,64, 128,64);
    convW(f(s0+4),  hGd1[d], 64,128, 64,128);
    convW(f(s0+6),  hGd2[d], 128,64, 128,64);
    convW(f(s0+8),  hU[d],   64,64, 64,64);
    convW(f(s0+10), hWu[d],  64,64, 64,64);
    convW(f(s0+12), hHu[d],  64,64, 64,64);
    convW(f(s0+14), hMs[d],  64,64, 64,64);
    convW(f(s0+16), hOut[d], 64,64, 64,64);
  }
  convW(f(99),  hPod1, 64,128, 64,128);
  convW(f(101), hPod2, 128,1,  128,16);

  // --- build inputs & lifting projections ---
  bno_pack_in<<<(NPIX+255)/256, 256, 0, stream>>>(sos, pml, src, kin);
  const int BIG = 1<<28;
  dense(kin,  hPxd1, f(4),  t128, NPIX, 128, 32, 1);
  dense(t128, hPxd2, f(6),  xb,   NPIX, 64, 128, 0);
  dense(kin,  hPkd1, f(8),  t128, NPIX, 128, 32, 1);
  dense(t128, hPkd2, f(10), kf,   NPIX, 64, 128, 0);

  // --- Born stages ---
  for (int d=0; d<4; ++d){
    int s0 = 11 + d*22;
    int act = (d < 3) ? 1 : 0;

    // L = project(k); u = dense(x); Ms = dense(k); s = L*u + Ms
    dense(kf,   hLd1[d], f(s0+1), t128, NPIX, 128, 64, act);
    dense(t128, hLd2[d], f(s0+3), b1,   NPIX, 64, 128, 0);
    dense(xb,   hU[d],   f(s0+9), b2,   NPIX, 64, 64, 0);
    dense(kf,   hMs[d],  f(s0+15),b3,   NPIX, 64, 64, 0);
    bno_ew_s<<<(NELT/4)/256, 256, 0, stream>>>((const float4*)b1, (const float4*)b2,
                                               (float4*)b3, NELT/4);

    // ---- spectral conv: b3 -> b1 (truncated DFT GEMMs) ----
    // F1: rfft along w, keep j<32.  rows (bh,c) M=32768, K=256(w), N=32(j)
    gemm(b3, fwr, nullptr, X1r, 32768, 32, 256, 64, 512,
         1,16384,0,64,  32,2048,0,1,  1.0f, 0, 0, BIG);
    gemm(b3, fwi, nullptr, X1i, 32768, 32, 256, 64, 512,
         1,16384,0,64,  32,2048,0,1,  1.0f, 0, 0, BIG);
    // F2: fft along h, keep 64 mode rows.  rows (b,c,j) M=4096, K=256(h), N=64(m')
    gemm(X1r, fhr, nullptr, X2r, 4096, 64, 256, 32, 64,
         1,32,524288,2048,  128,1,64,4096,  1.0f, 0, 0, BIG);
    gemm(X1i, fhi, nullptr, X2r, 4096, 64, 256, 32, 64,
         1,32,524288,2048,  128,1,64,4096, -1.0f, 1, 0, BIG);
    gemm(X1r, fhi, nullptr, X2i, 4096, 64, 256, 32, 64,
         1,32,524288,2048,  128,1,64,4096,  1.0f, 0, 0, BIG);
    gemm(X1i, fhr, nullptr, X2i, 4096, 64, 256, 32, 64,
         1,32,524288,2048,  128,1,64,4096,  1.0f, 1, 0, BIG);
    // per-mode complex 64x64 channel mix (k1 low block, k2 high block)
    bno_modemul_wmma<<<dim3(4,1,1024), 32, 0, stream>>>(X2r, X2i, f(s0+18), f(s0+19),
                                                        KKr, KKi, 0);
    bno_modemul_wmma<<<dim3(4,1,1024), 32, 0, stream>>>(X2r, X2i, f(s0+20), f(s0+21),
                                                        KKr, KKi, 1024);
    // inverse along h: rows (b,c,j) M=4096, K=64(m'), N=256(h)
    gemm(KKr, ghr, nullptr, Y1r, 4096, 256, 64, 32, 64,
         128,1,64,4096,  1,32,524288,2048,  1.0f, 0, 0, BIG);
    gemm(KKi, ghi, nullptr, Y1r, 4096, 256, 64, 32, 64,
         128,1,64,4096,  1,32,524288,2048, -1.0f, 1, 0, BIG);
    gemm(KKr, ghi, nullptr, Y1i, 4096, 256, 64, 32, 64,
         128,1,64,4096,  1,32,524288,2048,  1.0f, 0, 0, BIG);
    gemm(KKi, ghr, nullptr, Y1i, 4096, 256, 64, 32, 64,
         128,1,64,4096,  1,32,524288,2048,  1.0f, 1, 0, BIG);
    // inverse rfft along w (hermitian): rows (bh,c) M=32768, K=32(j), N=256(w)
    gemm(Y1r, gwr, nullptr, b1, 32768, 256, 32, 64, 512,
         32,2048,0,1,  1,16384,0,64,  1.0f, 0, 0, BIG);
    gemm(Y1i, gwi, nullptr, b1, 32768, 256, 32, 64, 512,
         32,2048,0,1,  1,16384,0,64, -1.0f, 1, 0, BIG);

    // G = project(k); Wu = dense(u); Hu = dense(u)
    dense(kf,   hGd1[d], f(s0+5), t128, NPIX, 128, 64, act);
    dense(t128, hGd2[d], f(s0+7), b4,   NPIX, 64, 128, 0);
    dense(b2,   hWu[d],  f(s0+11),b5,   NPIX, 64, 64, 0);
    dense(b2,   hHu[d],  f(s0+13),b3,   NPIX, 64, 64, 0);

    // born = Wu - G*(Hu - spec);  x = act(dense(born, out))
    bno_ew_born<<<(NELT/4)/256, 256, 0, stream>>>((const float4*)b5, (const float4*)b4,
                                                  (const float4*)b3, (const float4*)b1,
                                                  (float4*)b2, NELT/4);
    dense(b2, hOut[d], f(s0+17), xb, NPIX, 64, 64, act);
  }

  // --- proj_out on cropped region: rows (b,h<224,w<224), M = 100352 ---
  gemm(xb, hPod1, f(100), t128, 100352, 128, 64, 224, 224,
       64,16384,4194304,1,  128,28672,6422528,1,  1.0f, 0, 1, BIG);
  gemm(t128, hPod2, f(102), (float*)d_out, 100352, 16, 128, 100352, 1,
       128,0,0,1,  1,0,0,1,  1.0f, 0, 0, /*nMax=*/1);
}
